// QLSTM_65481071403564
// MI455X (gfx1250) — matrix operand order repaired and checked by reading, not compiled, not run on hardware
//
#include <hip/hip_runtime.h>
#include <hip/hip_bf16.h>
#include <math.h>

typedef _Float16 half_t;
typedef __attribute__((ext_vector_type(8)))  _Float16 v8h;
typedef __attribute__((ext_vector_type(16))) _Float16 v16h;
typedef __attribute__((ext_vector_type(8)))  float    v8f;

#define T_STEPS 256
#define BATCH   256
#define DMODEL  512
#define HDIM    512
#define ROWS    (T_STEPS * BATCH)   // 65536 independent encoder rows
#define SCAN_BLOCKS 64              // persistent-scan grid size (device-wide barrier count)

static __device__ __forceinline__ v16h cat8(v8h lo, v8h hi) {
  return __builtin_shufflevector(lo, hi, 0,1,2,3,4,5,6,7,8,9,10,11,12,13,14,15);
}
// A fragment: per ISA 16-bit A 16x32 layout, lane holds K chunks {+0..7, +16..23}
static __device__ __forceinline__ v16h ldA16(const half_t* p) {
  return cat8(*(const v8h*)p, *(const v8h*)(p + 16));
}
// B fragment: lane holds K chunks {+0..7, +16..23} (same as A)
static __device__ __forceinline__ v16h ldB16(const half_t* p) {
  return cat8(*(const v8h*)p, *(const v8h*)(p + 16));
}

// ---------------------------------------------------------------------------
// Sampler mask: value-invariant across timesteps; computed once on device.
// ---------------------------------------------------------------------------
__global__ void sampler_mask_kernel(const float* __restrict__ w1, const float* __restrict__ b1,
                                    const float* __restrict__ w2, const float* __restrict__ b2,
                                    float* __restrict__ mask) {
  if (threadIdx.x == 0 && blockIdx.x == 0) {
    float z1[8];
    for (int j = 0; j < 8; ++j) {
      float s = b1[j];
      for (int i = 0; i < 4; ++i) s += w1[i * 8 + j];   // ones(1,4) @ w1
      z1[j] = tanhf(s);
    }
    float z2[4];
    float mx = -1e30f;
    for (int j = 0; j < 4; ++j) {
      float s = b2[j];
      for (int i = 0; i < 8; ++i) s += z1[i] * w2[i * 4 + j];
      z2[j] = s;
      mx = fmaxf(mx, s);
    }
    float den = 0.f;
    for (int j = 0; j < 4; ++j) { z2[j] = expf(z2[j] - mx); den += z2[j]; }
    for (int j = 0; j < 4; ++j) mask[j] = z2[j] / den;
  }
}

// ---------------------------------------------------------------------------
// Weight transpose + f32->f16 convert: W[K,N] row-major -> WT[N,K] row-major.
// ---------------------------------------------------------------------------
__global__ void transpose_convert_kernel(const float* __restrict__ src, half_t* __restrict__ dst,
                                         int K, int N) {
  int idx = blockIdx.x * blockDim.x + threadIdx.x;
  if (idx < K * N) {
    int k = idx / N, n = idx - k * N;
    dst[(size_t)n * K + k] = (half_t)src[idx];
  }
}

__global__ void init_state_kernel(half_t* __restrict__ hx0, unsigned* __restrict__ sync_ctr) {
  int idx = blockIdx.x * blockDim.x + threadIdx.x;
  hx0[idx] = (half_t)0.f;
  if (idx == 0) *sync_ctr = 0u;
}

// ---------------------------------------------------------------------------
// Row-parallel GEMM, software-pipelined: out[M,N] = act(A @ W + bias), f16 out.
// One wave per 16x16 tile; K-loop via v_wmma_f32_16x16x32_f16 with next-iter
// fragments prefetched before the current WMMA issues.
// ---------------------------------------------------------------------------
template <bool A_F32, bool RELU>
__global__ __launch_bounds__(256) void gemm_wmma_f16out(const void* __restrict__ Asrc,
                                                        const half_t* __restrict__ WT,
                                                        const float* __restrict__ bias,
                                                        half_t* __restrict__ out,
                                                        int N, int K) {
  const int lane  = threadIdx.x & 31;
  const int wv    = threadIdx.x >> 5;
  const int tile  = blockIdx.x * 8 + wv;       // grid sized so every tile valid
  const int Nt    = N >> 4;
  const int mt    = tile / Nt;
  const int nt    = tile - mt * Nt;
  const int r0    = lane & 15;
  const int koffA = (lane >> 4) << 3;
  const int koffB = (lane >> 4) << 3;

  const size_t  arow  = (size_t)(mt * 16 + r0) * K;
  const half_t* bbase = WT + (size_t)(nt * 16 + r0) * K + koffB;

  v8f acc = {};
  v16h a_c, b_c;
  // prologue: fragments for k = 0
  if (A_F32) {
    const float* ap = (const float*)Asrc + arow + koffA;
    v16h t;
#pragma unroll
    for (int i = 0; i < 8; ++i) t[i] = (_Float16)ap[i];
#pragma unroll
    for (int i = 0; i < 8; ++i) t[8 + i] = (_Float16)ap[16 + i];
    a_c = t;
  } else {
    a_c = ldA16((const half_t*)Asrc + arow + koffA);
  }
  b_c = ldB16(bbase);

#pragma unroll 4
  for (int k = 0; k < K; k += 32) {
    const int kn = k + 32;
    v16h a_n, b_n;
    if (kn < K) {   // prefetch next iteration before issuing the current WMMA
      if (A_F32) {
        const float* ap = (const float*)Asrc + arow + kn + koffA;
        v16h t;
#pragma unroll
        for (int i = 0; i < 8; ++i) t[i] = (_Float16)ap[i];
#pragma unroll
        for (int i = 0; i < 8; ++i) t[8 + i] = (_Float16)ap[16 + i];
        a_n = t;
      } else {
        a_n = ldA16((const half_t*)Asrc + arow + kn + koffA);
      }
      b_n = ldB16(bbase + kn);
    }
    acc = __builtin_amdgcn_wmma_f32_16x16x32_f16(false, a_c, false, b_c, (short)0, acc, false, false);
    a_c = a_n;
    b_c = b_n;
  }

  const int   n     = nt * 16 + r0;
  const float bn    = bias[n];
  const int   mbase = mt * 16 + ((lane >> 4) << 3);
#pragma unroll
  for (int r = 0; r < 8; ++r) {
    float v = acc[r] + bn;
    if (RELU) v = fmaxf(v, 0.f);
    out[(size_t)(mbase + r) * N + n] = (half_t)v;
  }
}

// ---------------------------------------------------------------------------
// Persistent LSTM scan: ONE launch runs all 256 timesteps; blocks synchronize
// with a device-wide barrier (monotonic counter) between steps.
//   gates[B,4H] = [comp_t | hx] @ gate_w + gate_b   (K = 1024, f16 WMMA)
// Each wave owns one 16(batch) x 16(h) tile for all t, so cx lives entirely
// in VGPRs for the whole scan, and its 128KB slice of gate weights stays hot
// in cache. hx double-buffered in f16; grid barrier makes the recurrence safe.
// ---------------------------------------------------------------------------
__global__ __launch_bounds__(256) void lstm_scan_kernel(const half_t* __restrict__ comp,
                                                        half_t* __restrict__ hxA,
                                                        half_t* __restrict__ hxB,
                                                        const half_t* __restrict__ gWT,   // [4H,1024]
                                                        const float* __restrict__ gb,     // [4H]
                                                        const float* __restrict__ mask,   // [4]
                                                        float* __restrict__ out_stacked,  // [T,B,H]
                                                        float* __restrict__ out_hx,       // [B,H]
                                                        float* __restrict__ out_cx,       // [B,H]
                                                        unsigned* __restrict__ sync_ctr) {
  const int lane  = threadIdx.x & 31;
  const int wv    = threadIdx.x >> 5;
  const int tile  = blockIdx.x * 8 + wv;   // 512 tiles (64 blocks x 8 waves)
  const int mt    = tile >> 5;             // 16 batch tiles
  const int nt    = tile & 31;             // 32 h tiles
  const int r0    = lane & 15;
  const int koffA = (lane >> 4) << 3;
  const int koffB = (lane >> 4) << 3;

  const int n = nt * 16 + r0;
  const half_t* B0 = gWT + (size_t)(0 * HDIM + n) * 1024 + koffB;  // f
  const half_t* B1 = gWT + (size_t)(1 * HDIM + n) * 1024 + koffB;  // i
  const half_t* B2 = gWT + (size_t)(2 * HDIM + n) * 1024 + koffB;  // g
  const half_t* B3 = gWT + (size_t)(3 * HDIM + n) * 1024 + koffB;  // o

  const float m0 = mask[0], m1 = mask[1], m2 = mask[2], m3 = mask[3];
  const float bf = gb[n], bi = gb[HDIM + n], bg = gb[2 * HDIM + n], bo = gb[3 * HDIM + n];

  const half_t* AcBase = comp + (size_t)(mt * 16 + r0) * DMODEL + koffA;
  const size_t  ahOff  = (size_t)(mt * 16 + r0) * HDIM + koffA;
  const int     mbase  = mt * 16 + ((lane >> 4) << 3);

  float cxr[8];
#pragma unroll
  for (int r = 0; r < 8; ++r) cxr[r] = 0.f;

  for (int t = 0; t < T_STEPS; ++t) {
    const half_t* Ac = AcBase + (size_t)t * BATCH * DMODEL;
    const half_t* Ah = ((t & 1) ? hxB : hxA) + ahOff;
    half_t*       Ho = (t & 1) ? hxA : hxB;

    v8f accF = {}, accI = {}, accG = {}, accO = {};
    // prologue (k = 0 always reads comp)
    v16h a_c  = ldA16(Ac);
    v16h bF_c = ldB16(B0), bI_c = ldB16(B1), bG_c = ldB16(B2), bO_c = ldB16(B3);

#pragma unroll 4
    for (int k = 0; k < 1024; k += 32) {
      const int kn = k + 32;
      v16h a_n, bF_n, bI_n, bG_n, bO_n;
      if (kn < 1024) {   // prefetch next-iteration fragments first
        const half_t* apn = (kn < DMODEL) ? (Ac + kn) : (Ah + (kn - DMODEL));
        a_n  = ldA16(apn);
        bF_n = ldB16(B0 + kn);
        bI_n = ldB16(B1 + kn);
        bG_n = ldB16(B2 + kn);
        bO_n = ldB16(B3 + kn);
      }
      accF = __builtin_amdgcn_wmma_f32_16x16x32_f16(false, a_c, false, bF_c, (short)0, accF, false, false);
      accI = __builtin_amdgcn_wmma_f32_16x16x32_f16(false, a_c, false, bI_c, (short)0, accI, false, false);
      accG = __builtin_amdgcn_wmma_f32_16x16x32_f16(false, a_c, false, bG_c, (short)0, accG, false, false);
      accO = __builtin_amdgcn_wmma_f32_16x16x32_f16(false, a_c, false, bO_c, (short)0, accO, false, false);
      a_c = a_n; bF_c = bF_n; bI_c = bI_n; bG_c = bG_n; bO_c = bO_n;
    }

#pragma unroll
    for (int r = 0; r < 8; ++r) {
      const int    b   = mbase + r;
      const size_t idx = (size_t)b * HDIM + n;
      float f = m0 / (1.f + expf(-(accF[r] + bf)));
      float i = m1 / (1.f + expf(-(accI[r] + bi)));
      float g = m2 * tanhf(accG[r] + bg);
      float o = m3 / (1.f + expf(-(accO[r] + bo)));
      float c = f * cxr[r] + i * g;
      cxr[r] = c;                      // cell state stays in VGPRs for all 256 steps
      float h = o * tanhf(c);
      Ho[idx] = (half_t)h;
      out_stacked[((size_t)t * BATCH + b) * HDIM + n] = h;
      if (t == T_STEPS - 1) { out_hx[idx] = h; out_cx[idx] = c; }
    }

    // -------- device-wide barrier between timesteps (monotonic counter) ----
    __threadfence();                   // make this thread's stores agent-visible
    __syncthreads();                   // all waves in block done + fenced
    if (threadIdx.x == 0) {
      __hip_atomic_fetch_add(sync_ctr, 1u, __ATOMIC_RELEASE, __HIP_MEMORY_SCOPE_AGENT);
      const unsigned target = (unsigned)SCAN_BLOCKS * (unsigned)(t + 1);
      while (__hip_atomic_load(sync_ctr, __ATOMIC_ACQUIRE, __HIP_MEMORY_SCOPE_AGENT) < target) {
        __builtin_amdgcn_s_sleep(2);
      }
    }
    __syncthreads();
  }
}

// ---------------------------------------------------------------------------
extern "C" void kernel_launch(void* const* d_in, const int* in_sizes, int n_in,
                              void* d_out, int out_size, void* d_ws, size_t ws_size,
                              hipStream_t stream) {
  (void)in_sizes; (void)n_in; (void)out_size; (void)ws_size;

  const float* x       = (const float*)d_in[0];   // [T,B,512]
  const float* enc_w1  = (const float*)d_in[1];   // [512,128]
  const float* enc_b1  = (const float*)d_in[2];
  const float* enc_w2  = (const float*)d_in[3];   // [128,64]
  const float* enc_b2  = (const float*)d_in[4];
  const float* enc_w3  = (const float*)d_in[5];   // [64,512]
  const float* enc_b3  = (const float*)d_in[6];
  const float* gate_w  = (const float*)d_in[7];   // [1024,2048]
  const float* gate_b  = (const float*)d_in[8];
  const float* samp_w1 = (const float*)d_in[9];
  const float* samp_b1 = (const float*)d_in[10];
  const float* samp_w2 = (const float*)d_in[11];
  const float* samp_b2 = (const float*)d_in[12];

  char* ws = (char*)d_ws;
  size_t off = 0;
  auto take = [&](size_t bytes) -> char* {
    char* p = ws + off;
    off = (off + bytes + 255) & ~(size_t)255;
    return p;
  };
  float*    mask = (float*)   take(4 * sizeof(float));
  unsigned* sync = (unsigned*)take(sizeof(unsigned));
  half_t*   w1T  = (half_t*)  take((size_t)128  * 512  * 2);
  half_t*   w2T  = (half_t*)  take((size_t)64   * 128  * 2);
  half_t*   w3T  = (half_t*)  take((size_t)512  * 64   * 2);
  half_t*   gWT  = (half_t*)  take((size_t)2048 * 1024 * 2);
  half_t*   hxA  = (half_t*)  take((size_t)BATCH * HDIM * 2);
  half_t*   hxB  = (half_t*)  take((size_t)BATCH * HDIM * 2);
  half_t*   h1   = (half_t*)  take((size_t)ROWS * 128 * 2);
  half_t*   h2   = (half_t*)  take((size_t)ROWS * 64  * 2);
  half_t*   comp = (half_t*)  take((size_t)ROWS * 512 * 2);

  float* out_stacked = (float*)d_out;
  float* out_hx = out_stacked + (size_t)T_STEPS * BATCH * HDIM;
  float* out_cx = out_hx + (size_t)BATCH * HDIM;

  // Setup: mask, weight transpose/convert, state + barrier-counter init.
  sampler_mask_kernel<<<1, 64, 0, stream>>>(samp_w1, samp_b1, samp_w2, samp_b2, mask);
  transpose_convert_kernel<<<(512 * 128) / 256, 256, 0, stream>>>(enc_w1, w1T, 512, 128);
  transpose_convert_kernel<<<(128 * 64) / 256, 256, 0, stream>>>(enc_w2, w2T, 128, 64);
  transpose_convert_kernel<<<(64 * 512) / 256, 256, 0, stream>>>(enc_w3, w3T, 64, 512);
  transpose_convert_kernel<<<(1024 * 2048) / 256, 256, 0, stream>>>(gate_w, gWT, 1024, 2048);
  init_state_kernel<<<(BATCH * HDIM) / 256, 256, 0, stream>>>(hxA, sync);

  // Parallel phase: encoder over all T*B rows (recurrence-independent).
  gemm_wmma_f16out<true,  true ><<<(ROWS / 16) * (128 / 16) / 8, 256, 0, stream>>>(x,  w1T, enc_b1, h1,   128, 512);
  gemm_wmma_f16out<false, true ><<<(ROWS / 16) * (64  / 16) / 8, 256, 0, stream>>>(h1, w2T, enc_b2, h2,   64,  128);
  gemm_wmma_f16out<false, false><<<(ROWS / 16) * (512 / 16) / 8, 256, 0, stream>>>(h2, w3T, enc_b3, comp, 512, 64);

  // Sequential phase: single persistent kernel, device-wide barrier per step.
  lstm_scan_kernel<<<SCAN_BLOCKS, 256, 0, stream>>>(comp, hxA, hxB, gWT, gate_b, mask,
                                                    out_stacked, out_hx, out_cx, sync);
}
